// PAM_61100204753563
// MI455X (gfx1250) — compile-verified, matrix operand order checked by reading, not run on hardware
//
#include <hip/hip_runtime.h>
#include <hip/hip_bf16.h>

// ---------------------------------------------------------------------------
// PAM attention for MI455X (gfx1250, wave32, WMMA).
//   B=8, C=512, MID=64, H=W=64, N=4096.
// Pipeline:
//   1) proj_kernel   : fused q/k/v 1x1-conv GEMMs (bf16 WMMA, fp32 accum)
//                      q,k -> [B][N][64] bf16 ; v -> [B][N][512] bf16
//   2) vtrans_kernel : v [B][N][512] -> [B][512][N] bf16 (LDS transpose)
//   3) stats_kernel  : per-row max + sum(exp) of S = q.k^T (WMMA + lane
//                      butterfly reductions)
//   4) attn_kernel   : flash-style P.V with precomputed stats; 32 query rows
//                      per block (halves V traffic vs 16; each V fragment
//                      feeds 2 WMMAs); P staged through LDS to convert
//                      D-layout -> A-fragment layout; output transposed
//                      through LDS; gamma*feat + x residual; global_prefetch
//                      hints for the next key block.
// Workspace layout (assumes ws_size >= ~73 MB):
//   [0,4M)    qbf   [4M,8M)  kbf   [8M,40M)  v_nc   [40M,72M) v_cn
//   [72M, +128K) stats_m   then stats_l (+128K)
// ---------------------------------------------------------------------------

#define BATCH 8
#define CCH   512
#define MIDC  64
#define NPOS  4096

typedef __attribute__((ext_vector_type(8)))  float   v8f;
typedef __attribute__((ext_vector_type(16))) __bf16  v16bf;
typedef __attribute__((ext_vector_type(8)))  __bf16  v8bf;

// A/B fragment load for 16-bit WMMA operands from a per-lane row pointer that
// already includes the K-chunk base and klo. Halves 0..7 are contiguous at p,
// halves 8..15 contiguous at p+16 (ISA 7.12.2): two 16B loads.
__device__ __forceinline__ v16bf load_frag(const __bf16* p) {
  v8bf lo = *(const v8bf*)(p);
  v8bf hi = *(const v8bf*)(p + 16);
  v16bf r;
#pragma unroll
  for (int i = 0; i < 8; ++i) { r[i] = lo[i]; r[i + 8] = hi[i]; }
  return r;
}

__device__ __forceinline__ v8f wmma_bf16(v16bf a, v16bf b, v8f c) {
  // (neg_a, A, neg_b, B, c_mod, C, reuse_a, reuse_b)
  return __builtin_amdgcn_wmma_f32_16x16x32_bf16(false, a, false, b,
                                                 (short)0, c, false, false);
}

// ---------------------------------------------------------------------------
// Kernel 1: fused projections. Grid (N/16, B), block 256 (8 waves).
// Combined output channels: 0..63 -> q, 64..127 -> k, 128..639 -> v.
// Each wave owns 5 of the 40 m-tiles; K loop over C=512 in steps of 32.
// ---------------------------------------------------------------------------
__global__ void __launch_bounds__(256)
proj_kernel(const float* __restrict__ x,
            const float* __restrict__ wq, const float* __restrict__ bq,
            const float* __restrict__ wk, const float* __restrict__ bk,
            const float* __restrict__ wv, const float* __restrict__ bv,
            __bf16* __restrict__ qbf, __bf16* __restrict__ kbf,
            __bf16* __restrict__ vnc) {
  const int b    = blockIdx.y;
  const int nt   = blockIdx.x;
  const int lane = threadIdx.x & 31;
  const int w    = threadIdx.x >> 5;
  const int lh   = lane >> 4;
  const int klo  = lh * 8;
  const int li   = lane & 15;
  const int n    = nt * 16 + li;
  const float* xb = x + (size_t)b * CCH * NPOS;

  v8f acc[5];
#pragma unroll
  for (int s = 0; s < 5; ++s)
#pragma unroll
    for (int r = 0; r < 8; ++r) acc[s][r] = 0.0f;

  for (int kc = 0; kc < CCH; kc += 32) {
    v16bf af;  // A: rows = n positions, K = input channel
#pragma unroll
    for (int i = 0; i < 8; ++i) {
      af[i]     = (__bf16)xb[(size_t)(kc + klo + i) * NPOS + n];
      af[i + 8] = (__bf16)xb[(size_t)(kc + 16 + klo + i) * NPOS + n];
    }
#pragma unroll
    for (int s = 0; s < 5; ++s) {
      const int mcol = (w + 8 * s) * 16 + li;
      const float* wrow;
      if (mcol < 64)        wrow = wq + (size_t)mcol * CCH;
      else if (mcol < 128)  wrow = wk + (size_t)(mcol - 64) * CCH;
      else                  wrow = wv + (size_t)(mcol - 128) * CCH;
      v16bf bfm;  // B: cols = output channel, K = input channel (contiguous)
#pragma unroll
      for (int i = 0; i < 8; ++i) {
        bfm[i]     = (__bf16)wrow[kc + klo + i];
        bfm[i + 8] = (__bf16)wrow[kc + 16 + klo + i];
      }
      acc[s] = wmma_bf16(af, bfm, acc[s]);
    }
  }

#pragma unroll
  for (int s = 0; s < 5; ++s) {
    const int mcol = (w + 8 * s) * 16 + li;
    float bias; __bf16* dst; int ld;
    if (mcol < 64)       { bias = bq[mcol];       dst = qbf + (size_t)b * NPOS * MIDC + mcol;        ld = MIDC; }
    else if (mcol < 128) { bias = bk[mcol - 64];  dst = kbf + (size_t)b * NPOS * MIDC + (mcol - 64); ld = MIDC; }
    else                 { bias = bv[mcol - 128]; dst = vnc + (size_t)b * NPOS * CCH  + (mcol - 128);ld = CCH;  }
#pragma unroll
    for (int r = 0; r < 8; ++r) {
      const int nr = nt * 16 + r + 8 * lh;   // D row
      dst[(size_t)nr * ld] = (__bf16)(acc[s][r] + bias);
    }
  }
}

// ---------------------------------------------------------------------------
// Kernel 2: transpose v [B][N][512] -> [B][512][N] via 64x64 LDS tiles.
// Grid (N/64, C/64, B), block 256. All global traffic in 16B chunks.
// ---------------------------------------------------------------------------
__global__ void __launch_bounds__(256)
vtrans_kernel(const __bf16* __restrict__ vnc, __bf16* __restrict__ vcn) {
  const int b  = blockIdx.z;
  const int n0 = blockIdx.x * 64;
  const int c0 = blockIdx.y * 64;
  __shared__ __bf16 tile[64][72];  // +8 pad to spread banks
  const int t   = threadIdx.x;
  const int rr  = t >> 3;
  const int cc8 = (t & 7) * 8;
  const __bf16* src = vnc + ((size_t)b * NPOS + n0) * CCH + c0;
#pragma unroll
  for (int p = 0; p < 2; ++p) {
    const int row = rr + 32 * p;
    v8bf v = *(const v8bf*)(src + (size_t)row * CCH + cc8);
#pragma unroll
    for (int i = 0; i < 8; ++i) tile[row][cc8 + i] = v[i];
  }
  __syncthreads();
  __bf16* dst = vcn + ((size_t)b * CCH + c0) * NPOS + n0;
#pragma unroll
  for (int p = 0; p < 2; ++p) {
    const int crow = rr + 32 * p;
    v8bf v;
#pragma unroll
    for (int i = 0; i < 8; ++i) v[i] = tile[cc8 + i][crow];
    *(v8bf*)(dst + (size_t)crow * NPOS + cc8) = v;
  }
}

// ---------------------------------------------------------------------------
// Kernel 3: softmax row stats (max, sum of exp) via WMMA score recompute.
// Grid (N/16/8, B), block 256; each wave handles one 16-row query tile.
// ---------------------------------------------------------------------------
__global__ void __launch_bounds__(256)
stats_kernel(const __bf16* __restrict__ qbf, const __bf16* __restrict__ kbf,
             float* __restrict__ stats_m, float* __restrict__ stats_l) {
  const int b    = blockIdx.y;
  const int w    = threadIdx.x >> 5;
  const int nt   = blockIdx.x * 8 + w;
  const int lane = threadIdx.x & 31;
  const int lh   = lane >> 4;
  const int klo  = lh * 8;
  const int li   = lane & 15;

  const __bf16* qrow = qbf + ((size_t)b * NPOS + nt * 16 + li) * MIDC;
  v16bf qa0 = load_frag(qrow + klo);
  v16bf qa1 = load_frag(qrow + 32 + klo);

  float m[8], l[8];
#pragma unroll
  for (int r = 0; r < 8; ++r) { m[r] = -3.0e38f; l[r] = 0.0f; }

  for (int jt = 0; jt < NPOS; jt += 16) {
    const __bf16* krow = kbf + ((size_t)b * NPOS + jt + li) * MIDC;
    if (jt + 16 < NPOS)
      __builtin_prefetch(krow + 16 * MIDC, 0, 1);  // global_prefetch_b8
    v16bf kb0 = load_frag(krow + klo);
    v16bf kb1 = load_frag(krow + 32 + klo);
    v8f s;
#pragma unroll
    for (int r = 0; r < 8; ++r) s[r] = 0.0f;
    s = wmma_bf16(qa0, kb0, s);
    s = wmma_bf16(qa1, kb1, s);
#pragma unroll
    for (int r = 0; r < 8; ++r) {
      float tm = s[r];  // row (r+8*lh), 16 lanes of this half hold 16 cols
      tm = fmaxf(tm, __shfl_xor(tm, 1, 32));
      tm = fmaxf(tm, __shfl_xor(tm, 2, 32));
      tm = fmaxf(tm, __shfl_xor(tm, 4, 32));
      tm = fmaxf(tm, __shfl_xor(tm, 8, 32));
      const float mn = fmaxf(m[r], tm);
      float e = __expf(s[r] - mn);
      e += __shfl_xor(e, 1, 32);
      e += __shfl_xor(e, 2, 32);
      e += __shfl_xor(e, 4, 32);
      e += __shfl_xor(e, 8, 32);
      l[r] = l[r] * __expf(m[r] - mn) + e;
      m[r] = mn;
    }
  }
  if (li == 0) {
#pragma unroll
    for (int r = 0; r < 8; ++r) {
      const size_t idx = (size_t)b * NPOS + nt * 16 + r + 8 * lh;
      stats_m[idx] = m[r];
      stats_l[idx] = l[r];
    }
  }
}

// ---------------------------------------------------------------------------
// Kernel 4: fused P.V with residual. Grid (N/32, B), block 256 (8 waves).
// 32 query rows/block (two 16-row halves); wave w owns dv cols [64w,64w+64)
// (4 accum tiles per row-half = 8 total). Key loop of 128:
//   - wave computes its two 16x16 score subtiles (4 WMMAs), exps them against
//     precomputed row max, stages bf16 probabilities in LDS [32][128],
//   - then 32 P.V WMMAs: each V B-fragment (2x16B loads) feeds both
//     row-half accumulators -> 1 fragment load per WMMA.
// Final: scale 1/l, LDS transpose (two 16-row passes), gamma*feat + x.
// ---------------------------------------------------------------------------
#define PSTR 136  // 128 + 8 halves pad; 272B rows keep 16B alignment
#define QT   32

__global__ void __launch_bounds__(256)
attn_kernel(const __bf16* __restrict__ qbf, const __bf16* __restrict__ kbf,
            const __bf16* __restrict__ vcn,
            const float* __restrict__ stats_m, const float* __restrict__ stats_l,
            const float* __restrict__ x, const float* __restrict__ gamma,
            float* __restrict__ out) {
  const int b    = blockIdx.y;
  const int nt   = blockIdx.x;        // 32-row tile index
  const int n0   = nt * QT;
  const int w    = threadIdx.x >> 5;
  const int lane = threadIdx.x & 31;
  const int lh   = lane >> 4;
  const int klo  = lh * 8;
  const int li   = lane & 15;

  __shared__ __bf16 Pbuf[QT * PSTR];
  __shared__ float  stage[CCH * 17];

  // Q fragments for both 16-row halves.
  v16bf qa[2][2];
  float m_r[2][8], linv[2][8];
#pragma unroll
  for (int h = 0; h < 2; ++h) {
    const __bf16* qrow = qbf + ((size_t)b * NPOS + n0 + h * 16 + li) * MIDC;
    qa[h][0] = load_frag(qrow + klo);
    qa[h][1] = load_frag(qrow + 32 + klo);
#pragma unroll
    for (int r = 0; r < 8; ++r) {
      const size_t idx = (size_t)b * NPOS + n0 + h * 16 + r + 8 * lh;
      m_r[h][r]  = stats_m[idx];
      linv[h][r] = 1.0f / stats_l[idx];
    }
  }

  v8f acc[2][4];
#pragma unroll
  for (int h = 0; h < 2; ++h)
#pragma unroll
    for (int t = 0; t < 4; ++t)
#pragma unroll
      for (int r = 0; r < 8; ++r) acc[h][t][r] = 0.0f;

  for (int kb = 0; kb < NPOS; kb += 128) {
    // --- scores for this wave's 16 keys, both row-halves ---
    const int j = kb + w * 16 + li;
    const __bf16* krow = kbf + ((size_t)b * NPOS + j) * MIDC;
    if (kb + 128 < NPOS) {
      __builtin_prefetch(krow + 128 * MIDC, 0, 1);               // next K rows
      __builtin_prefetch(vcn + ((size_t)b * CCH + w * 64 + li) * NPOS
                             + kb + 128, 0, 1);                  // next V rows
    }
    v16bf kf0 = load_frag(krow + klo);
    v16bf kf1 = load_frag(krow + 32 + klo);
#pragma unroll
    for (int h = 0; h < 2; ++h) {
      v8f s;
#pragma unroll
      for (int r = 0; r < 8; ++r) s[r] = 0.0f;
      s = wmma_bf16(qa[h][0], kf0, s);
      s = wmma_bf16(qa[h][1], kf1, s);
      // unnormalized probabilities -> LDS (row-major [32][128])
#pragma unroll
      for (int r = 0; r < 8; ++r) {
        const float p = __expf(s[r] - m_r[h][r]);
        Pbuf[(h * 16 + r + 8 * lh) * PSTR + w * 16 + li] = (__bf16)p;
      }
    }
    __syncthreads();
    // --- P.V: A-frags from LDS, B-frags from v [B][C][N] (K contiguous);
    //     each V fragment is reused by both row-half accumulators. ---
#pragma unroll
    for (int c4 = 0; c4 < 4; ++c4) {
      v16bf pa0 = load_frag(&Pbuf[li * PSTR + c4 * 32 + klo]);
      v16bf pa1 = load_frag(&Pbuf[(16 + li) * PSTR + c4 * 32 + klo]);
#pragma unroll
      for (int t = 0; t < 4; ++t) {
        const int c = w * 64 + t * 16 + li;
        const __bf16* vrow = vcn + ((size_t)b * CCH + c) * NPOS + kb + c4 * 32;
        v16bf vf = load_frag(vrow + klo);
        acc[0][t] = wmma_bf16(pa0, vf, acc[0][t]);
        acc[1][t] = wmma_bf16(pa1, vf, acc[1][t]);
      }
    }
    __syncthreads();
  }

  // --- normalize, transpose through LDS (two 16-row passes), residual ---
  const float g = gamma[0];
#pragma unroll
  for (int h = 0; h < 2; ++h) {
#pragma unroll
    for (int t = 0; t < 4; ++t) {
      const int c = w * 64 + t * 16 + li;
#pragma unroll
      for (int r = 0; r < 8; ++r)
        stage[c * 17 + r + 8 * lh] = acc[h][t][r] * linv[h][r];
    }
    __syncthreads();
#pragma unroll
    for (int cc = 0; cc < 2; ++cc) {
      const int c = threadIdx.x * 2 + cc;
      const size_t base = ((size_t)b * CCH + c) * NPOS + n0 + h * 16;
#pragma unroll
      for (int nl = 0; nl < 16; ++nl)
        out[base + nl] = g * stage[c * 17 + nl] + x[base + nl];
    }
    __syncthreads();  // stage is reused by the next row-half
  }
}

// ---------------------------------------------------------------------------
extern "C" void kernel_launch(void* const* d_in, const int* in_sizes, int n_in,
                              void* d_out, int out_size, void* d_ws, size_t ws_size,
                              hipStream_t stream) {
  const float* x     = (const float*)d_in[0];
  const float* wq    = (const float*)d_in[1];
  const float* bq    = (const float*)d_in[2];
  const float* wk    = (const float*)d_in[3];
  const float* bk    = (const float*)d_in[4];
  const float* wv    = (const float*)d_in[5];
  const float* bv    = (const float*)d_in[6];
  const float* gamma = (const float*)d_in[7];
  float* out = (float*)d_out;

  char* ws = (char*)d_ws;                                   // needs ~73 MB
  __bf16* qbf = (__bf16*)(ws);                              // 4 MB
  __bf16* kbf = (__bf16*)(ws + (4ull  << 20));              // 4 MB
  __bf16* vnc = (__bf16*)(ws + (8ull  << 20));              // 32 MB
  __bf16* vcn = (__bf16*)(ws + (40ull << 20));              // 32 MB
  float* stats_m = (float*)(ws + (72ull << 20));            // 128 KB
  float* stats_l = (float*)(ws + (72ull << 20) + (128ull << 10));

  proj_kernel<<<dim3(NPOS / 16, BATCH), 256, 0, stream>>>(
      x, wq, bq, wk, bk, wv, bv, qbf, kbf, vnc);
  vtrans_kernel<<<dim3(NPOS / 64, CCH / 64, BATCH), 256, 0, stream>>>(vnc, vcn);
  stats_kernel<<<dim3(NPOS / 16 / 8, BATCH), 256, 0, stream>>>(
      qbf, kbf, stats_m, stats_l);
  attn_kernel<<<dim3(NPOS / 32, BATCH), 256, 0, stream>>>(
      qbf, kbf, vcn, stats_m, stats_l, x, gamma, out);
}
